// PNA_68109591380382
// MI455X (gfx1250) — matrix operand order repaired and checked by reading, not compiled, not run on hardware
//
#include <hip/hip_runtime.h>
#include <math.h>

// ---------- types ----------
typedef unsigned short u16;
typedef __attribute__((ext_vector_type(16))) __bf16 v16bf;
typedef __attribute__((ext_vector_type(8)))  float  v8f;

struct alignas(16) U128 { unsigned int x[4]; };
union V16 { v16bf v; u16 s[16]; U128 q[2]; };

// ---------- helpers ----------
__device__ __forceinline__ u16 f2b(float f) {
    unsigned u = __builtin_bit_cast(unsigned, f);
    unsigned r = u + 0x7FFFu + ((u >> 16) & 1u);   // round-to-nearest-even
    return (u16)(r >> 16);
}
__device__ __forceinline__ float b2f(u16 s) {
    unsigned u = ((unsigned)s) << 16;
    return __builtin_bit_cast(float, u);
}
// order-preserving float<->uint key (larger float => larger key)
__device__ __forceinline__ unsigned fkey(float f) {
    unsigned u = __builtin_bit_cast(unsigned, f);
    return (u & 0x80000000u) ? ~u : (u | 0x80000000u);
}
__device__ __forceinline__ float funkey(unsigned k) {
    unsigned u = (k & 0x80000000u) ? (k ^ 0x80000000u) : ~k;
    return __builtin_bit_cast(float, u);
}

__device__ __forceinline__ v8f wmma_bf16(v16bf a, v16bf b, v8f c) {
    return __builtin_amdgcn_wmma_f32_16x16x32_bf16(false, a, false, b, (short)0, c, false, false);
}

// A fragment from a row-major LDS bf16 row (16-bit A layout: lane holds two
// contiguous 8-element K runs at kbase and kbase+16; kbase includes the
// half-wave offset of 8).
__device__ __forceinline__ v16bf load_frag_a(const u16* row, int kbase) {
    V16 u;
    u.q[0] = *(const U128*)(row + kbase);
    u.q[1] = *(const U128*)(row + kbase + 16);
    return u.v;
}
__device__ __forceinline__ v16bf load_frag_a_scaled(const u16* row, int kbase, float s) {
    V16 u;
    u.q[0] = *(const U128*)(row + kbase);
    u.q[1] = *(const U128*)(row + kbase + 16);
#pragma unroll
    for (int i = 0; i < 16; ++i) u.s[i] = f2b(b2f(u.s[i]) * s);
    return u.v;
}
// B fragment from pre-packed weights: 32 bytes contiguous per lane per fragment
__device__ __forceinline__ v16bf load_frag_b(const u16* packed, int frag, int lane) {
    V16 u;
    const U128* p = (const U128*)(packed + ((frag * 32 + lane) << 4));
    u.q[0] = p[0];
    u.q[1] = p[1];
    return u.v;
}

// ---------- utility kernels ----------
__global__ void fill_u32_kernel(unsigned* p, unsigned v, int count) {
    int i = blockIdx.x * blockDim.x + threadIdx.x;
    if (i < count) p[i] = v;
}
__global__ void cvt_bf16_kernel(const float* __restrict__ in, u16* __restrict__ out, int count) {
    int i = blockIdx.x * blockDim.x + threadIdx.x;
    if (i < count) out[i] = f2b(in[i]);
}
// Pack a row-major f32 [K][N] weight into WMMA-B-fragment order (bf16).
// Fragment (kc,nc): lane L holds 16 contiguous bf16; element i maps to
// K = kc*32 + (L>=16 ? 8 : 0) + (i&7) + (i>=8 ? 16 : 0), N = nc*16 + (L&15)  (ISA B layout).
__global__ void pack_b_kernel(const float* __restrict__ W, u16* __restrict__ P, int K, int N) {
    int tid = blockIdx.x * blockDim.x + threadIdx.x;
    int total = K * N;
    if (tid >= total) return;
    int frag = tid >> 9;
    int rem  = tid & 511;
    int lane = rem >> 4;
    int i    = rem & 15;
    int nch  = N >> 4;
    int kc = frag / nch, nc = frag % nch;
    int k = kc * 32 + ((lane >> 4) << 3) + (i & 7) + ((i >> 3) << 4);
    int n = nc * 16 + (lane & 15);
    P[tid] = f2b(W[k * N + n]);
}

__global__ void deg_kernel(const int* __restrict__ dst, float* __restrict__ deg, int E) {
    int e = blockIdx.x * blockDim.x + threadIdx.x;
    if (e < E) atomicAdd(&deg[dst[e]], 1.0f);
}
__global__ void deg_stats_kernel(const float* __restrict__ deg, float* __restrict__ stats, int n) {
    __shared__ float s0[256], s1[256];
    int tid = threadIdx.x;
    float a = 0.f, b = 0.f;
    for (int i = blockIdx.x * blockDim.x + tid; i < n; i += gridDim.x * blockDim.x) {
        float d = deg[i];
        a += d;
        b += logf(d + 1.0f);
    }
    s0[tid] = a; s1[tid] = b;
    __syncthreads();
    for (int off = 128; off; off >>= 1) {
        if (tid < off) { s0[tid] += s0[tid + off]; s1[tid] += s1[tid + off]; }
        __syncthreads();
    }
    if (tid == 0) { atomicAdd(&stats[0], s0[0]); atomicAdd(&stats[1], s1[0]); }
}
__global__ void scalers_kernel(const float* __restrict__ deg, const float* __restrict__ stats,
                               float* __restrict__ amp, float* __restrict__ att,
                               float* __restrict__ lsc, int n) {
    int i = blockIdx.x * blockDim.x + threadIdx.x;
    if (i >= n) return;
    float avg_lin = stats[0] / (float)n;
    float avg_log = stats[1] / (float)n;
    float d  = deg[i];
    float lg = logf(d + 1.0f);
    amp[i] = lg / avg_log;
    att[i] = avg_log / logf(fmaxf(d, 1.0f) + 1.0f);
    lsc[i] = d / avg_lin;
}

// ---------- edge message kernel (persistent waves, register-resident weights) ----------
// h[e] = cat(x[dst[e]], x[src[e]]) @ preW(128x64) + preB ; scatter 4 stats to dst.
// The 128x64 bf16 weight = 16 B-fragments = 128 VGPRs/lane: loaded ONCE per wave,
// then a grid-stride loop processes edge tiles with only LDS A-loads + WMMA +
// atomics in the hot loop. __launch_bounds__(128,1) lifts the VGPR cap so the
// fragments stay register-resident (no scratch spills). Tiles are wave-private
// (DS ops are in-order per wave), so waves run fully independently.
__global__ void __launch_bounds__(128, 1)
edge_msg_kernel(const u16* __restrict__ xb,          // [n][64] bf16
                const int* __restrict__ src, const int* __restrict__ dst,
                const u16* __restrict__ preWp,       // packed bf16 B frags
                const float* __restrict__ preB,
                float* __restrict__ sum, float* __restrict__ sumsq,
                unsigned* __restrict__ mnk, unsigned* __restrict__ mxk,
                int nEdges) {
    __shared__ __align__(16) u16 xt[4][16][136];  // 272B row stride: 16B aligned + bank-staggered
    __shared__ int idxs[4][16];
    const int wave = threadIdx.x >> 5;
    const int lane = threadIdx.x & 31;
    const int nt   = (nEdges + 15) >> 4;
    const int nWaves = gridDim.x * 4;
    const int wid  = blockIdx.x * 4 + wave;
    const int rc   = lane & 15;
    const int part = lane >> 4;                     // 0: dst half, 1: src half
    const int off8 = part << 3;

    // ---- one-time per wave: weights + bias into registers ----
    v16bf bw[16];
#pragma unroll
    for (int f = 0; f < 16; ++f) bw[f] = load_frag_b(preWp, f, lane);
    float bias[4];
#pragma unroll
    for (int nc = 0; nc < 4; ++nc) bias[nc] = preB[nc * 16 + rc];

    for (int tile = wid; tile < nt; tile += nWaves) {
        // ---- gather: 16 rows of cat(x[dst], x[src]) into this wave's LDS tile ----
        {
            int e = tile * 16 + rc;
            if (e >= nEdges) e = nEdges - 1;
            int node = part ? src[e] : dst[e];
            if (part == 0) idxs[wave][rc] = node;
            const U128* g = (const U128*)(xb + (size_t)node * 64);
            U128* l = (U128*)(&xt[wave][rc][part * 64]);
#pragma unroll
            for (int j = 0; j < 8; ++j) l[j] = g[j];   // 128B coalesced row copy
        }
        __builtin_amdgcn_wave_barrier();   // keep DS store->load program order

        // ---- 16x64 tile GEMM, bias folded into C init ----
        const u16* row = &xt[wave][rc][0];
        v8f c4[4];
#pragma unroll
        for (int nc = 0; nc < 4; ++nc) {
            v8f c;
#pragma unroll
            for (int i = 0; i < 8; ++i) c[i] = bias[nc];
            c4[nc] = c;
        }
#pragma unroll
        for (int kc = 0; kc < 4; ++kc) {
            v16bf a = load_frag_a(row, kc * 32 + off8);
#pragma unroll
            for (int nc = 0; nc < 4; ++nc) c4[nc] = wmma_bf16(a, bw[kc * 4 + nc], c4[nc]);
        }

        // ---- scatter 4 stats per element (32-bit offsets -> saddr+voffset atomics) ----
        int didx[8];
#pragma unroll
        for (int r8 = 0; r8 < 8; ++r8) didx[r8] = idxs[wave][r8 + off8];
        __builtin_amdgcn_wave_barrier();   // idx loads before next iteration's overwrite

        const bool full = (tile * 16 + 16) <= nEdges;   // uniform; true for exact tiles
#pragma unroll
        for (int r8 = 0; r8 < 8; ++r8) {
            if (full || (tile * 16 + r8 + off8) < nEdges) {
                int base = didx[r8] * 64 + rc;
#pragma unroll
                for (int nc = 0; nc < 4; ++nc) {
                    float v  = c4[nc][r8];
                    int  o   = base + nc * 16;
                    atomicAdd(&sum[o], v);
                    atomicAdd(&sumsq[o], v * v);
                    unsigned k = fkey(v);
                    atomicMax(&mxk[o], k);
                    atomicMin(&mnk[o], k);
                }
            }
        }
    }
}

// ---------- node post-MLP kernel ----------
// z = cat(x, agg, agg*amp, agg*att, agg*lin)  (K = 1088, never materialized)
// out = relu?( (z @ postW + postB) @ linW + linB )
template <int OUTF, bool RELU, bool OUT_BF16>
__global__ void node_post_kernel(const u16* __restrict__ xb,   // [n][64] bf16 (x or h1)
                                 const float* __restrict__ sum, const float* __restrict__ sumsq,
                                 const unsigned* __restrict__ mnk, const unsigned* __restrict__ mxk,
                                 const float* __restrict__ deg,
                                 const float* __restrict__ amp, const float* __restrict__ att,
                                 const float* __restrict__ lsc,
                                 const u16* __restrict__ postWp, const float* __restrict__ postB,
                                 const u16* __restrict__ linWp,  const float* __restrict__ linB,
                                 u16* __restrict__ outBf, float* __restrict__ outF,
                                 int nNodes) {
    __shared__ __align__(16) u16  aggT[2][16][264];  // mean|min|max|std (4*64 cols), staggered
    __shared__ __align__(16) u16  xT[2][16][72];
    __shared__ float scalT[2][16][3];
    __shared__ float outT[2][16][72];
    const int wave = threadIdx.x >> 5;
    const int lane = threadIdx.x & 31;
    const int tile = blockIdx.x * 2 + wave;
    const int nt   = (nNodes + 15) >> 4;
    const bool active = tile < nt;
    const bool full  = (tile * 16 + 16) <= nNodes;   // uniform per wave

    if (active) {
        for (int idx = lane; idx < 1024; idx += 32) {
            int r = idx >> 6, f = idx & 63;
            int node = tile * 16 + r;
            float mean = 0.f, mnv = 0.f, mxv = 0.f, sd = 0.f;
            u16 xv = 0;
            if (full || node < nNodes) {
                int a = node * 64 + f;
                float d = deg[node];
                float safe = fmaxf(d, 1.0f);
                float s = sum[a];
                float q = sumsq[a];
                mean = s / safe;
                float var = q / safe - mean * mean;
                if (var < 0.f) var = 0.f;
                sd = sqrtf(var + 1e-5f);
                bool has = d > 0.f;
                mnv = has ? funkey(mnk[a]) : 0.f;
                mxv = has ? funkey(mxk[a]) : 0.f;
                xv = xb[a];
            }
            aggT[wave][r][f]       = f2b(mean);
            aggT[wave][r][64 + f]  = f2b(mnv);
            aggT[wave][r][128 + f] = f2b(mxv);
            aggT[wave][r][192 + f] = f2b(sd);
            xT[wave][r][f] = xv;
        }
        if (lane < 16) {
            int node = tile * 16 + lane;
            float a = 1.f, b = 1.f, c = 1.f;
            if (full || node < nNodes) { a = amp[node]; b = att[node]; c = lsc[node]; }
            scalT[wave][lane][0] = a;
            scalT[wave][lane][1] = b;
            scalT[wave][lane][2] = c;
        }
    }
    __syncthreads();

    const int rc   = lane & 15;
    const int off8 = (lane >> 4) << 3;
    const int NCH  = OUTF / 16;

    if (active) {
        for (int nc = 0; nc < NCH; ++nc) {
            v8f c;
            const float bias = postB[nc * 16 + rc];
#pragma unroll
            for (int i = 0; i < 8; ++i) c[i] = bias;
#pragma unroll 2
            for (int kc = 0; kc < 34; ++kc) {
                int kstart = kc * 32;
                v16bf a;
                if (kstart < 64) {                       // x block
                    a = load_frag_a(&xT[wave][rc][0], kstart + off8);
                } else {                                  // agg blocks (raw / *amp / *att / *lin)
                    int t = kstart - 64;
                    int sec = t >> 8;
                    int acol = (t & 255) + off8;
                    if (sec == 0) a = load_frag_a(&aggT[wave][rc][0], acol);
                    else          a = load_frag_a_scaled(&aggT[wave][rc][0], acol,
                                                         scalT[wave][rc][sec - 1]);
                }
                v16bf b = load_frag_b(postWp, kc * NCH + nc, lane);
                c = wmma_bf16(a, b, c);
            }
#pragma unroll
            for (int r8 = 0; r8 < 8; ++r8) outT[wave][r8 + off8][nc * 16 + rc] = c[r8];
        }
    }
    __syncthreads();

    if (active) {
        const int KCH = OUTF / 32;
        for (int nc = 0; nc < NCH; ++nc) {
            v8f c;
            const float bias = linB[nc * 16 + rc];
#pragma unroll
            for (int i = 0; i < 8; ++i) c[i] = bias;
#pragma unroll
            for (int kc = 0; kc < KCH; ++kc) {
                V16 u;
                int kbase = kc * 32 + off8;
#pragma unroll
                for (int i = 0; i < 8; ++i) u.s[i]     = f2b(outT[wave][rc][kbase + i]);
#pragma unroll
                for (int i = 0; i < 8; ++i) u.s[8 + i] = f2b(outT[wave][rc][kbase + 16 + i]);
                v16bf b = load_frag_b(linWp, kc * NCH + nc, lane);
                c = wmma_bf16(u.v, b, c);
            }
#pragma unroll
            for (int r8 = 0; r8 < 8; ++r8) {
                int node = tile * 16 + r8 + off8;
                if (full || node < nNodes) {
                    float v = c[r8];
                    if (RELU) v = fmaxf(v, 0.0f);
                    int col = nc * 16 + rc;
                    if constexpr (OUT_BF16) outBf[node * OUTF + col] = f2b(v);
                    else                    outF [node * OUTF + col] = v;
                }
            }
        }
    }
}

// ---------- log_softmax over 32 classes, one wave per node ----------
__global__ void logsoftmax_kernel(const float* __restrict__ in, float* __restrict__ out, int n) {
    int wave = threadIdx.x >> 5, lane = threadIdx.x & 31;
    int node = blockIdx.x * (blockDim.x >> 5) + wave;
    if (node >= n) return;
    float v = in[node * 32 + lane];
    float m = v;
#pragma unroll
    for (int off = 16; off; off >>= 1) m = fmaxf(m, __shfl_xor(m, off, 32));
    float s = expf(v - m);
#pragma unroll
    for (int off = 16; off; off >>= 1) s += __shfl_xor(s, off, 32);
    out[node * 32 + lane] = v - m - logf(s);
}

// ---------- launcher ----------
extern "C" void kernel_launch(void* const* d_in, const int* in_sizes, int n_in,
                              void* d_out, int out_size, void* d_ws, size_t ws_size,
                              hipStream_t stream) {
    const int F = 64, HID = 64, CLS = 32;
    const int N = in_sizes[0] / F;
    const int E = in_sizes[1] / 2;

    const float* x      = (const float*)d_in[0];
    const int*   ei     = (const int*)d_in[1];
    const int*   src    = ei;
    const int*   dst    = ei + E;
    const float* preW1  = (const float*)d_in[2];
    const float* preB1  = (const float*)d_in[3];
    const float* postW1 = (const float*)d_in[4];
    const float* postB1 = (const float*)d_in[5];
    const float* linW1  = (const float*)d_in[6];
    const float* linB1  = (const float*)d_in[7];
    const float* preW2  = (const float*)d_in[8];
    const float* preB2  = (const float*)d_in[9];
    const float* postW2 = (const float*)d_in[10];
    const float* postB2 = (const float*)d_in[11];
    const float* linW2  = (const float*)d_in[12];
    const float* linB2  = (const float*)d_in[13];

    char* w = (char*)d_ws;
    size_t off = 0;
    auto alloc = [&](size_t bytes) -> void* {
        off = (off + 255) & ~(size_t)255;
        void* p = w + off;
        off += bytes;
        return p;
    };

    const int NF = N * 64;
    float*    deg    = (float*)alloc(sizeof(float) * N);
    float*    stats  = (float*)alloc(sizeof(float) * 2);
    float*    amp    = (float*)alloc(sizeof(float) * N);
    float*    att    = (float*)alloc(sizeof(float) * N);
    float*    lsc    = (float*)alloc(sizeof(float) * N);
    u16*      xb     = (u16*)alloc(sizeof(u16) * (size_t)NF);
    u16*      h1b    = (u16*)alloc(sizeof(u16) * (size_t)NF);
    float*    sum    = (float*)alloc(sizeof(float) * (size_t)NF);
    float*    sumsq  = (float*)alloc(sizeof(float) * (size_t)NF);
    unsigned* mnk    = (unsigned*)alloc(sizeof(unsigned) * (size_t)NF);
    unsigned* mxk    = (unsigned*)alloc(sizeof(unsigned) * (size_t)NF);
    u16*      preW1p = (u16*)alloc(sizeof(u16) * 128 * F);
    u16*      preW2p = (u16*)alloc(sizeof(u16) * 128 * HID);
    u16*      postW1p= (u16*)alloc(sizeof(u16) * 1088 * HID);
    u16*      postW2p= (u16*)alloc(sizeof(u16) * 1088 * CLS);
    u16*      linW1p = (u16*)alloc(sizeof(u16) * HID * HID);
    u16*      linW2p = (u16*)alloc(sizeof(u16) * CLS * CLS);
    float*    out2   = (float*)alloc(sizeof(float) * (size_t)N * CLS);

    auto fill = [&](void* p, unsigned v, int cnt) {
        fill_u32_kernel<<<(cnt + 255) / 256, 256, 0, stream>>>((unsigned*)p, v, cnt);
    };

    // --- degrees and scalers ---
    fill(deg, 0u, N);
    fill(stats, 0u, 2);
    deg_kernel<<<(E + 255) / 256, 256, 0, stream>>>(dst, deg, E);
    deg_stats_kernel<<<128, 256, 0, stream>>>(deg, stats, N);
    scalers_kernel<<<(N + 255) / 256, 256, 0, stream>>>(deg, stats, amp, att, lsc, N);

    // --- precision prep ---
    cvt_bf16_kernel<<<(NF + 255) / 256, 256, 0, stream>>>(x, xb, NF);
    pack_b_kernel<<<(128 * F + 255) / 256, 256, 0, stream>>>(preW1, preW1p, 128, F);
    pack_b_kernel<<<(128 * HID + 255) / 256, 256, 0, stream>>>(preW2, preW2p, 128, HID);
    pack_b_kernel<<<(1088 * HID + 255) / 256, 256, 0, stream>>>(postW1, postW1p, 1088, HID);
    pack_b_kernel<<<(1088 * CLS + 255) / 256, 256, 0, stream>>>(postW2, postW2p, 1088, CLS);
    pack_b_kernel<<<(HID * HID + 255) / 256, 256, 0, stream>>>(linW1, linW1p, HID, HID);
    pack_b_kernel<<<(CLS * CLS + 255) / 256, 256, 0, stream>>>(linW2, linW2p, CLS, CLS);

    const unsigned KEY_PINF = 0xFF800000u;  // fkey(+inf) -> init for atomicMin
    const unsigned KEY_NINF = 0x007FFFFFu;  // fkey(-inf) -> init for atomicMax
    const int eTiles = (E + 15) / 16;
    int eBlocks = (eTiles + 3) / 4;
    if (eBlocks > 2048) eBlocks = 2048;     // persistent waves: ~6 tiles/wave at E=800k
    const int nTiles = (N + 15) / 16;
    const int nBlocks = (nTiles + 1) / 2;

    // --- layer 1 ---
    fill(sum, 0u, NF); fill(sumsq, 0u, NF); fill(mnk, KEY_PINF, NF); fill(mxk, KEY_NINF, NF);
    edge_msg_kernel<<<eBlocks, 128, 0, stream>>>(xb, src, dst, preW1p, preB1,
                                                 sum, sumsq, mnk, mxk, E);
    node_post_kernel<64, true, true><<<nBlocks, 64, 0, stream>>>(
        xb, sum, sumsq, mnk, mxk, deg, amp, att, lsc,
        postW1p, postB1, linW1p, linB1, h1b, nullptr, N);

    // --- layer 2 ---
    fill(sum, 0u, NF); fill(sumsq, 0u, NF); fill(mnk, KEY_PINF, NF); fill(mxk, KEY_NINF, NF);
    edge_msg_kernel<<<eBlocks, 128, 0, stream>>>(h1b, src, dst, preW2p, preB2,
                                                 sum, sumsq, mnk, mxk, E);
    node_post_kernel<32, false, false><<<nBlocks, 64, 0, stream>>>(
        h1b, sum, sumsq, mnk, mxk, deg, amp, att, lsc,
        postW2p, postB2, linW2p, linB2, nullptr, out2, N);

    // --- log softmax ---
    logsoftmax_kernel<<<(N + 3) / 4, 128, 0, stream>>>(out2, (float*)d_out, N);
    (void)ws_size; (void)n_in; (void)out_size;
}
